// Gate_2697239462625
// MI455X (gfx1250) — matrix operand order repaired and checked by reading, not compiled, not run on hardware
//
#include <hip/hip_runtime.h>
#include <hip/hip_bf16.h>

// ---------------------------------------------------------------------------
// MoE gate (DeepSeek-style group-limited routing) for MI455X / gfx1250.
//   Phase 1: convert weight fp32 -> bf16 (3.7 MB, stays resident in 192MB L2)
//   Phase 2: logits = x @ W^T via v_wmma_f32_16x16x32_bf16, fp32 accum,
//            software-pipelined K loop with double-buffered loads
//   Phase 3: per-token routing (sigmoid, group top-2, top-4 groups, top-8)
// ---------------------------------------------------------------------------

#define DIM        7168
#define NEXP       256
#define NGROUPS    8
#define GROUP_SZ   (NEXP / NGROUPS)   // 32
#define TOPKG      4
#define TOPK       8
#define NTOK       16384
#define ROUTE_SCALE 2.5f
#define MTILE      64                 // tokens per block (4 x 16 per wave)
#define KSTEP      32                 // K per WMMA

typedef __attribute__((ext_vector_type(16))) __bf16 v16bf;
typedef __attribute__((ext_vector_type(4)))  __bf16 v4bf;
typedef __attribute__((ext_vector_type(8)))  float  v8f;
typedef __attribute__((ext_vector_type(4)))  float  v4f;
typedef __attribute__((ext_vector_type(8)))  unsigned v8u;
typedef __attribute__((ext_vector_type(2)))  unsigned v2u;

static_assert(DIM % (2 * KSTEP) == 0, "K tiling (double-buffered)");
static_assert(NEXP == 16 * 16, "16 waves x 16 experts");

// ---------------------------------------------------------------------------
// Phase 1: weight fp32 -> bf16 (row-major [NEXP, DIM])
// ---------------------------------------------------------------------------
__global__ void gate_cvt_weight(const float* __restrict__ w,
                                __bf16* __restrict__ wb, int n4) {
  int i = blockIdx.x * blockDim.x + threadIdx.x;
  if (i >= n4) return;
  int idx = i * 4;
  v4f f = *(const v4f*)(w + idx);
  v4bf o = __builtin_convertvector(f, v4bf);   // 2x v_cvt_pk_bf16_f32
  *(v4bf*)(wb + idx) = o;
}

// ---------------------------------------------------------------------------
// Phase 2 helpers
// ---------------------------------------------------------------------------
// Issue all global loads for one K-step: B fragment (bf16, direct) and the
// raw fp32 A data for 4 M-tiles (converted later, off the load critical path).
__device__ __forceinline__ void load_tile(const float* __restrict__ aptr,
                                          const __bf16* __restrict__ bptr,
                                          int k0, v4f (&r)[4][4], v16bf& b) {
  b = *(const v16bf*)(bptr + k0);
  #pragma unroll
  for (int m = 0; m < 4; ++m) {
    const float* ap = aptr + (size_t)(m * 16) * DIM + k0;
    r[m][0] = *(const v4f*)(ap + 0);
    r[m][1] = *(const v4f*)(ap + 4);
    r[m][2] = *(const v4f*)(ap + 8);
    r[m][3] = *(const v4f*)(ap + 12);
  }
}

// Pack 16 staged fp32 values into one bf16 WMMA A fragment using pairwise
// packed converts (v_cvt_pk_bf16_f32) and whole-VGPR inserts only (no
// half-register v_mov_b16 assembly).
__device__ __forceinline__ v16bf cvt_frag(const v4f (&r)[4]) {
  v8u au;
  #pragma unroll
  for (int j = 0; j < 4; ++j) {
    v4bf q = __builtin_convertvector(r[j], v4bf);  // 2x v_cvt_pk_bf16_f32
    v2u qu = __builtin_bit_cast(v2u, q);
    au[2 * j + 0] = qu[0];
    au[2 * j + 1] = qu[1];
  }
  return __builtin_bit_cast(v16bf, au);
}

// Convert the staged fp32 data into 4 distinct bf16 A fragments, then issue
// the 4 WMMAs back-to-back (distinct A/acc registers -> no intra-step WAR).
__device__ __forceinline__ void mma_tile(const v4f (&r)[4][4], const v16bf& b,
                                         v8f (&acc)[4]) {
  v16bf a[4];
  #pragma unroll
  for (int m = 0; m < 4; ++m) a[m] = cvt_frag(r[m]);
  #pragma unroll
  for (int m = 0; m < 4; ++m) {
    acc[m] = __builtin_amdgcn_wmma_f32_16x16x32_bf16(
        false, a[m], false, b, (short)0, acc[m], false, false);
  }
}

// ---------------------------------------------------------------------------
// Phase 2: logits[T, NEXP] = x[T, DIM] @ W^T  (bf16 WMMA, f32 accumulate)
// Block: 512 threads = 16 waves; wave w covers experts [16w, 16w+16),
// 4 M-tiles of 16 tokens per wave (64 tokens per block).
// ---------------------------------------------------------------------------
__global__ __launch_bounds__(512)
void gate_gemm(const float* __restrict__ x,
               const __bf16* __restrict__ wb,
               float* __restrict__ logits) {
  const int lane    = threadIdx.x & 31;
  const int wave    = threadIdx.x >> 5;    // 0..15
  const int l15     = lane & 15;
  const int lhalf   = lane >> 4;           // 0 | 1
  const int nBase   = wave * 16;
  const int tokBase = blockIdx.x * MTILE;

  v8f acc[4];
  #pragma unroll
  for (int m = 0; m < 4; ++m) acc[m] = (v8f)0.0f;

  // B: lane holds expert row (nBase + l15), 16 consecutive K (lhalf selects
  //    K-halves 0..15 / 16..31 of the KSTEP window) -> contiguous bf16 load.
  const __bf16* bptr = wb + (size_t)(nBase + l15) * DIM + lhalf * 16;
  // A: lane<16 holds K {+0..15}; lane>=16 holds K {+16..31}.
  const float* aptr = x + (size_t)(tokBase + l15) * DIM + lhalf * 16;

  // Software pipeline: double-buffered raw loads, converts+WMMAs one step
  // behind the loads so waits are staged and WMMA WAR distance is a full step.
  v4f r0[4][4], r1[4][4];
  v16bf b0, b1;
  load_tile(aptr, bptr, 0, r0, b0);

  int k0 = 0;
  #pragma unroll 1
  for (; k0 < DIM - 2 * KSTEP; k0 += 2 * KSTEP) {
    load_tile(aptr, bptr, k0 + KSTEP, r1, b1);
    mma_tile(r0, b0, acc);
    load_tile(aptr, bptr, k0 + 2 * KSTEP, r0, b0);
    mma_tile(r1, b1, acc);
  }
  // epilogue (k0 == DIM - 64)
  load_tile(aptr, bptr, k0 + KSTEP, r1, b1);
  mma_tile(r0, b0, acc);
  mma_tile(r1, b1, acc);

  // C/D layout: lanes 0-15 -> N=lane, M=vgpr; lanes 16-31 -> N=lane-16, M=8+vgpr
  float* outCol = logits + (size_t)tokBase * NEXP + (nBase + l15);
  #pragma unroll
  for (int m = 0; m < 4; ++m) {
    #pragma unroll
    for (int r = 0; r < 8; ++r) {
      int row = m * 16 + lhalf * 8 + r;
      outCol[(size_t)row * NEXP] = acc[m][r];
    }
  }
}

// ---------------------------------------------------------------------------
// Phase 3: per-token routing. One thread per token.
// ---------------------------------------------------------------------------
__device__ __forceinline__ float sigmoidf(float v) {
  return 1.0f / (1.0f + __expf(-v));
}

__global__ __launch_bounds__(256)
void gate_route(const float* __restrict__ logits,
                const float* __restrict__ bias,
                const float* __restrict__ ebias,
                float* __restrict__ outW,
                int* __restrict__ outI) {
  int t = blockIdx.x * blockDim.x + threadIdx.x;
  if (t >= NTOK) return;
  const float* row = logits + (size_t)t * NEXP;
  const float NEG = -__builtin_inff();

  // group score = sum of top-2 routing scores within each group of 32
  float gs[NGROUPS];
  #pragma unroll
  for (int g = 0; g < NGROUPS; ++g) {
    float m1 = NEG, m2 = NEG;
    for (int j = 0; j < GROUP_SZ; ++j) {
      int e = g * GROUP_SZ + j;
      float s = sigmoidf(row[e]) + bias[e] + ebias[e];
      if (s > m1) { m2 = m1; m1 = s; }
      else if (s > m2) { m2 = s; }
    }
    gs[g] = m1 + m2;
  }

  // top-4 groups (strict > keeps lowest index on ties, like jax top_k)
  unsigned gmask = 0;
  #pragma unroll
  for (int i = 0; i < TOPKG; ++i) {
    int best = 0; float bv = NEG;
    #pragma unroll
    for (int g = 0; g < NGROUPS; ++g) {
      bool taken = (gmask >> g) & 1u;
      if (!taken && gs[g] > bv) { bv = gs[g]; best = g; }
    }
    gmask |= 1u << best;
  }

  // top-8 experts among selected groups (sorted-desc insertion list)
  float tv[TOPK]; int ti[TOPK];
  #pragma unroll
  for (int i = 0; i < TOPK; ++i) { tv[i] = NEG; ti[i] = 0; }
  for (int g = 0; g < NGROUPS; ++g) {
    if (!((gmask >> g) & 1u)) continue;
    for (int j = 0; j < GROUP_SZ; ++j) {
      int e = g * GROUP_SZ + j;
      float s = sigmoidf(row[e]) + bias[e] + ebias[e];
      if (s > tv[TOPK - 1]) {
        int p = TOPK - 1;
        #pragma unroll
        for (int q = TOPK - 1; q > 0; --q) {
          if (tv[q - 1] < s) { tv[q] = tv[q - 1]; ti[q] = ti[q - 1]; p = q - 1; }
        }
        tv[p] = s; ti[p] = e;
      }
    }
  }

  // gather original sigmoid scores, normalize, scale
  float wv[TOPK]; float wsum = 0.0f;
  #pragma unroll
  for (int i = 0; i < TOPK; ++i) { wv[i] = sigmoidf(row[ti[i]]); wsum += wv[i]; }
  float sc = ROUTE_SCALE / wsum;
  #pragma unroll
  for (int i = 0; i < TOPK; ++i) {
    outW[(size_t)t * TOPK + i] = wv[i] * sc;
    outI[(size_t)t * TOPK + i] = ti[i];
  }
}

// ---------------------------------------------------------------------------
extern "C" void kernel_launch(void* const* d_in, const int* in_sizes, int n_in,
                              void* d_out, int out_size, void* d_ws, size_t ws_size,
                              hipStream_t stream) {
  const float* x      = (const float*)d_in[0];
  const float* weight = (const float*)d_in[1];
  const float* bias   = (const float*)d_in[2];
  const float* ebias  = (const float*)d_in[3];

  float* outW = (float*)d_out;                     // [NTOK, TOPK] fp32
  int*   outI = (int*)d_out + (size_t)NTOK * TOPK; // [NTOK, TOPK] int32 bits

  // workspace: logits [NTOK, NEXP] fp32, then weight bf16 [NEXP, DIM]
  float*  logits = (float*)d_ws;
  __bf16* wb = (__bf16*)((char*)d_ws + (size_t)NTOK * NEXP * sizeof(float));

  // Phase 1: weight -> bf16
  {
    int n4 = (NEXP * DIM) / 4;
    int blk = 256;
    gate_cvt_weight<<<(n4 + blk - 1) / blk, blk, 0, stream>>>(weight, wb, n4);
  }
  // Phase 2: WMMA GEMM
  {
    dim3 grid(NTOK / MTILE);   // 256 blocks
    dim3 block(512);           // 16 waves
    gate_gemm<<<grid, block, 0, stream>>>(x, wb, logits);
  }
  // Phase 3: routing
  {
    int blk = 256;
    gate_route<<<(NTOK + blk - 1) / blk, blk, 0, stream>>>(logits, bias, ebias,
                                                           outW, outI);
  }
}
